// TripletLoss_33500744908931
// MI455X (gfx1250) — compile-verified, matrix operand order checked by reading, not run on hardware
//
#include <hip/hip_runtime.h>
#include <math.h>

// CDNA5 / gfx1250: wave32, WMMA 16x16x32 bf16 -> f32 accumulate.
typedef __attribute__((ext_vector_type(16))) __bf16 v16bf;
typedef __attribute__((ext_vector_type(8)))  __bf16 v8bf;
typedef __attribute__((ext_vector_type(8)))  float  v8f;

#define NN 64
#define TT 256
#define WW 64
#define EE 128
#define MARGIN 0.2f

// ---------------------------------------------------------------------------
// Pass 1: one-time fp32 -> (bf16 hi, bf16 lo) error-compensated split.
// Each thread converts 8 contiguous floats and stores two 16-byte bf16 chunks.
// ---------------------------------------------------------------------------
__global__ __launch_bounds__(256) void
split_kernel(const float* __restrict__ in, __bf16* __restrict__ hi,
             __bf16* __restrict__ lo, int n8) {
  int idx = blockIdx.x * blockDim.x + threadIdx.x;
  if (idx >= n8) return;
  const float4* p = (const float4*)in + (size_t)idx * 2;
  float4 f0 = p[0], f1 = p[1];
  float f[8] = {f0.x, f0.y, f0.z, f0.w, f1.x, f1.y, f1.z, f1.w};
  v8bf h, l;
#pragma unroll
  for (int j = 0; j < 8; ++j) {
    __bf16 hj = (__bf16)f[j];
    h[j] = hj;
    l[j] = (__bf16)(f[j] - (float)hj);
  }
  *(v8bf*)(hi + (size_t)idx * 8) = h;
  *(v8bf*)(lo + (size_t)idx * 8) = l;
}

// Fragment gather per CDNA5 16-bit A/B layout: lanes 0-15 hold row M=lane,
// K = {b..b+7, b+16..b+23}; lanes 16-31 hold K = {b+8..b+15, b+24..b+31}
// (caller pre-offsets p by +8 elements for lanes >= 16). Two b128 loads.
__device__ __forceinline__ v16bf load_frag(const __bf16* __restrict__ p) {
  v8bf a = *(const v8bf*)p;
  v8bf b = *(const v8bf*)(p + 16);
  return __builtin_shufflevector(a, b, 0, 1, 2, 3, 4, 5, 6, 7,
                                 8, 9, 10, 11, 12, 13, 14, 15);
}

// ---------------------------------------------------------------------------
// Pass 2: one block per (audio a = blockIdx.y, text i = blockIdx.x).
// 8 waves; wave w owns M-tiles {2w, 2w+1} x all 4 N-tiles of the 256x64
// alignment matrix. K=128 in 4 steps of 32; per step load A frags (2 M-tiles,
// hi+lo) and B frags (4 N-tiles, hi+lo) once, then 2*4*3 WMMAs.
// Split product: a*b ~= a_hi*b_hi + a_hi*b_lo + a_lo*b_hi (f32 accumulate).
// ---------------------------------------------------------------------------
__global__ __launch_bounds__(256) void
score_kernel(const __bf16* __restrict__ Ahi, const __bf16* __restrict__ Alo,
             const __bf16* __restrict__ Bhi, const __bf16* __restrict__ Blo,
             float* __restrict__ S) {
  const int i    = blockIdx.x;   // text index
  const int a    = blockIdx.y;   // audio index
  const int tid  = threadIdx.x;
  const int wave = tid >> 5;
  const int lane = tid & 31;
  const int row  = lane & 15;            // M (or N) row within 16-wide tile
  const int ksub = (lane >> 4) << 3;     // +0 for lanes 0-15, +8 for 16-31

  const size_t abase = (size_t)a * TT * EE;
  const size_t bbase = (size_t)i * WW * EE;

  const v8f vzero = {0.f, 0.f, 0.f, 0.f, 0.f, 0.f, 0.f, 0.f};
  v8f acc[2][4];
#pragma unroll
  for (int mi = 0; mi < 2; ++mi)
#pragma unroll
    for (int nt = 0; nt < 4; ++nt) acc[mi][nt] = vzero;

#pragma unroll
  for (int ks = 0; ks < 4; ++ks) {
    const int kofs = ks * 32 + ksub;
    v16bf ahi[2], alo[2];
#pragma unroll
    for (int mi = 0; mi < 2; ++mi) {
      const size_t off = abase + (size_t)((wave * 2 + mi) * 16 + row) * EE + kofs;
      ahi[mi] = load_frag(Ahi + off);
      alo[mi] = load_frag(Alo + off);
    }
    v16bf bhi[4], blo[4];
#pragma unroll
    for (int nt = 0; nt < 4; ++nt) {
      const size_t off = bbase + (size_t)(nt * 16 + row) * EE + kofs;
      bhi[nt] = load_frag(Bhi + off);
      blo[nt] = load_frag(Blo + off);
    }
#pragma unroll
    for (int mi = 0; mi < 2; ++mi)
#pragma unroll
      for (int nt = 0; nt < 4; ++nt) {
        acc[mi][nt] = __builtin_amdgcn_wmma_f32_16x16x32_bf16(
            false, ahi[mi], false, bhi[nt], (short)0, acc[mi][nt], false, false);
        acc[mi][nt] = __builtin_amdgcn_wmma_f32_16x16x32_bf16(
            false, ahi[mi], false, blo[nt], (short)0, acc[mi][nt], false, false);
        acc[mi][nt] = __builtin_amdgcn_wmma_f32_16x16x32_bf16(
            false, alo[mi], false, bhi[nt], (short)0, acc[mi][nt], false, false);
      }
  }

  // relu + per-lane sum over this wave's 2x4 16x16 output tiles
  float partial = 0.0f;
#pragma unroll
  for (int mi = 0; mi < 2; ++mi)
#pragma unroll
    for (int nt = 0; nt < 4; ++nt)
#pragma unroll
      for (int j = 0; j < 8; ++j)
        partial += fmaxf(acc[mi][nt][j], 0.0f);

  // Block reduction -> S[a,i] = sum / (T*W)
  __shared__ float red[256];
  red[tid] = partial;
  __syncthreads();
#pragma unroll
  for (int s = 128; s > 0; s >>= 1) {
    if (tid < s) red[tid] += red[tid + s];
    __syncthreads();
  }
  if (tid == 0) S[a * NN + i] = red[0] * (1.0f / (float)(TT * WW));
}

// ---------------------------------------------------------------------------
// Pass 3: hard-negative triplet loss on the 64x64 score matrix.
// ---------------------------------------------------------------------------
__global__ __launch_bounds__(64) void
loss_kernel(const float* __restrict__ S, const int* __restrict__ fids,
            float* __restrict__ out) {
  const int i = threadIdx.x;   // 0..63
  __shared__ float Ls[NN];

  const float sii = S[i * NN + i];
  const int   fi  = fids[i];
  float mcol = -__builtin_inff();  // hardest negative audio for text i
  float mrow = -__builtin_inff();  // hardest negative text for audio i
  for (int j = 0; j < NN; ++j) {
    if (fids[j] != fi) {
      mcol = fmaxf(mcol, S[j * NN + i]);  // S[a=j, i]
      mrow = fmaxf(mrow, S[i * NN + j]);  // S[a=i, t=j]
    }
  }
  Ls[i] = fmaxf(mcol - sii + MARGIN, 0.0f) + fmaxf(mrow - sii + MARGIN, 0.0f);
  __syncthreads();
  if (i == 0) {
    float s = 0.0f;
    for (int j = 0; j < NN; ++j) s += Ls[j];
    *out = s * (1.0f / (float)NN);
  }
}

extern "C" void kernel_launch(void* const* d_in, const int* in_sizes, int n_in,
                              void* d_out, int out_size, void* d_ws, size_t ws_size,
                              hipStream_t stream) {
  const float* audio = (const float*)d_in[0];  // (64, 256, 128) f32
  const float* text  = (const float*)d_in[1];  // (64, 64, 128)  f32
  const int*   fids  = (const int*)d_in[2];    // (64,)          i32
  float* out = (float*)d_out;                  // scalar f32

  // Workspace layout (all 256B-aligned):
  //   [S 64*64 f32][audio_hi][audio_lo][text_hi][text_lo]  (bf16 arrays)
  const size_t nA = (size_t)NN * TT * EE;      // 2,097,152
  const size_t nB = (size_t)NN * WW * EE;      //   524,288
  char* ws = (char*)d_ws;
  float*  S   = (float*)ws;                    size_t off = 16384;
  __bf16* Ahi = (__bf16*)(ws + off);           off += nA * 2;
  __bf16* Alo = (__bf16*)(ws + off);           off += nA * 2;
  __bf16* Bhi = (__bf16*)(ws + off);           off += nB * 2;
  __bf16* Blo = (__bf16*)(ws + off);

  // Pass 1: one-time hi/lo bf16 split (grid-stride free: exact sizing).
  const int a8 = (int)(nA / 8), b8 = (int)(nB / 8);
  split_kernel<<<(a8 + 255) / 256, 256, 0, stream>>>(audio, Ahi, Alo, a8);
  split_kernel<<<(b8 + 255) / 256, 256, 0, stream>>>(text, Bhi, Blo, b8);

  // Pass 2: all-pairs relu-mean alignment scores via bf16 WMMA.
  dim3 grid(NN, NN);   // x = text i, y = audio a (same-a blocks adjacent -> L2 reuse)
  score_kernel<<<grid, 256, 0, stream>>>(Ahi, Alo, Bhi, Blo, S);

  // Pass 3: scalar loss.
  loss_kernel<<<1, NN, 0, stream>>>(S, fids, out);
}
